// OmniAnomaly_12764642804110
// MI455X (gfx1250) — compile-verified, matrix operand order checked by reading, not run on hardware
//
#include <hip/hip_runtime.h>
#include <hip/hip_bf16.h>
#include <math.h>

// Problem constants
#define Bsz 256
#define Wt_ 128
#define INd 128
#define Hd  1024
#define Zd  128
#define Dd  1024
#define OUTd 128
#define Kfl 3

typedef __attribute__((ext_vector_type(16))) __bf16 v16bf;
typedef __attribute__((ext_vector_type(8)))  __bf16 v8bf;
typedef __attribute__((ext_vector_type(8)))  float  v8f;

union FragBF { v16bf v; v8bf h[2]; };

__device__ __forceinline__ v8f wmma_bf16(v16bf a, v16bf b, v8f c) {
  // D = A(16x32 bf16) * B(32x16 bf16) + C(16x16 f32)
  return __builtin_amdgcn_wmma_f32_16x16x32_bf16(false, a, false, b, (short)0, c, false, false);
}

// B fragment: B[k][n] = W[row0+n][k]  (W row-major bf16, ldw elements per row)
__device__ __forceinline__ v16bf load_fragB_w(const __bf16* __restrict__ W, int ldw,
                                              int row0, int k0, int lane) {
  int n = lane & 15, half = lane >> 4;
  const __bf16* p = W + (long)(row0 + n) * ldw + k0 + half * 8;
  FragBF f;
  f.h[0] = *(const v8bf*)(p);
  f.h[1] = *(const v8bf*)(p + 16);
  return f.v;
}

// A fragment from LDS bf16 tile (16 rows x 32 k, row-major)
__device__ __forceinline__ v16bf load_fragA_lds(const __bf16* sA, int lane) {
  int m = lane & 15, half = lane >> 4;
  const __bf16* p = sA + m * 32 + half * 8;
  FragBF f;
  f.h[0] = *(const v8bf*)(p);
  f.h[1] = *(const v8bf*)(p + 16);
  return f.v;
}

// Stage a 16x32 bf16 tile (rows m0.., k-cols k0..k0+31) into LDS.
// 64 threads x 16B vector copy (global_load_b128 -> ds_store_b128);
// also prefetch the next k-chunk into cache.
__device__ __forceinline__ void stage_tile(__bf16* sA, const __bf16* __restrict__ A,
                                           long lda, long m0, int k0, int kend, int tid) {
  if (tid < 64) {
    int r = tid >> 2, c = tid & 3;
    const __bf16* src = A + (m0 + r) * lda + k0 + c * 8;
    *(v8bf*)(sA + r * 32 + c * 8) = *(const v8bf*)src;
    if (k0 + 32 < kend) __builtin_prefetch(src + 32, 0, 0);
  }
}

__device__ __forceinline__ float sigmoidf_(float x) { return 1.f / (1.f + __expf(-x)); }
__device__ __forceinline__ float softplusf_(float x) { return x > 20.f ? x : log1pf(__expf(x)); }

// ---------------------------------------------------------------------------
// fp32 -> bf16 conversion (weights + x)
// ---------------------------------------------------------------------------
__global__ void cvt_bf16_kernel(const float* __restrict__ s, __bf16* __restrict__ d, int n) {
  int i = blockIdx.x * 256 + threadIdx.x;
  if (i < n) d[i] = (__bf16)s[i];
}

// ---------------------------------------------------------------------------
// Fused GRU step: Hnext = GRUCell(Hprev, X), bf16 WMMA + LDS-staged A operand
// grid = (Bsz/16, Hd/128), block 256 (8 waves, one 16x16 tile each)
// ---------------------------------------------------------------------------
__global__ __launch_bounds__(256) void gru_step_kernel(
    const __bf16* __restrict__ Xb, long ldx,          // bf16 step input [Bsz x 128]
    const float*  __restrict__ Hprev,                 // f32 state (gate blend)
    const __bf16* __restrict__ Hprevb,                // bf16 state (GEMM operand)
    const __bf16* __restrict__ Wih,                   // [3*Hd x 128] bf16
    const __bf16* __restrict__ Whh,                   // [3*Hd x Hd] bf16
    const float* __restrict__ bih, const float* __restrict__ bhh,
    float* __restrict__ Hnext, __bf16* __restrict__ Hnextb)
{
  __shared__ __align__(16) __bf16 sA[16 * 32];
  const int tid = threadIdx.x, lane = tid & 31, wv = tid >> 5;
  const int m0 = blockIdx.x * 16;
  const int n0 = blockIdx.y * 128 + wv * 16;

  v8f ar{}, az{}, ain{}, ahn{};

  // ---- input contribution (K = 128) ----
  for (int k0 = 0; k0 < 128; k0 += 32) {
    __syncthreads();
    stage_tile(sA, Xb, ldx, m0, k0, 128, tid);
    __syncthreads();
    v16bf a = load_fragA_lds(sA, lane);
    ar  = wmma_bf16(a, load_fragB_w(Wih, 128, 0 * Hd + n0, k0, lane), ar);
    az  = wmma_bf16(a, load_fragB_w(Wih, 128, 1 * Hd + n0, k0, lane), az);
    ain = wmma_bf16(a, load_fragB_w(Wih, 128, 2 * Hd + n0, k0, lane), ain);
  }

  // ---- hidden contribution (K = 1024) ----
  for (int k0 = 0; k0 < Hd; k0 += 32) {
    __syncthreads();
    stage_tile(sA, Hprevb, Hd, m0, k0, Hd, tid);
    __syncthreads();
    v16bf a = load_fragA_lds(sA, lane);
    ar  = wmma_bf16(a, load_fragB_w(Whh, Hd, 0 * Hd + n0, k0, lane), ar);
    az  = wmma_bf16(a, load_fragB_w(Whh, Hd, 1 * Hd + n0, k0, lane), az);
    ahn = wmma_bf16(a, load_fragB_w(Whh, Hd, 2 * Hd + n0, k0, lane), ahn);
  }

  const int col = n0 + (lane & 15);
  const int half = lane >> 4;
  const float br  = bih[col]          + bhh[col];
  const float bz  = bih[Hd + col]     + bhh[Hd + col];
  const float bin = bih[2 * Hd + col];
  const float bhn = bhh[2 * Hd + col];
#pragma unroll
  for (int j = 0; j < 8; ++j) {
    long b = m0 + j + half * 8;
    float r  = sigmoidf_(ar[j] + br);
    float u  = sigmoidf_(az[j] + bz);
    float nn = tanhf(ain[j] + bin + r * (ahn[j] + bhn));
    float ho = Hprev[b * Hd + col];
    float hn = (1.f - u) * nn + u * ho;
    Hnext[b * Hd + col]  = hn;
    Hnextb[b * Hd + col] = (__bf16)hn;
  }
}

// ---------------------------------------------------------------------------
// Outb = [A1 | A2] @ W.T + bias  (bf16 in, bf16 out; W row-major [N x (K1+K2)])
// grid = (M/16, N/128), block 256. A1 may be null (K1 still offsets A2 cols).
// ---------------------------------------------------------------------------
__global__ __launch_bounds__(256) void gemm_bias_kernel(
    const __bf16* __restrict__ A1, long lda1, int K1,
    const __bf16* __restrict__ A2, long lda2, int K2,
    const __bf16* __restrict__ Wt, int ldw,
    const float* __restrict__ bias,
    __bf16* __restrict__ Outb, int ldo)
{
  __shared__ __align__(16) __bf16 sA[16 * 32];
  const int tid = threadIdx.x, lane = tid & 31, wv = tid >> 5;
  const long m0 = (long)blockIdx.x * 16;
  const int n0 = blockIdx.y * 128 + wv * 16;

  v8f acc{};
  if (A1) {
    for (int k0 = 0; k0 < K1; k0 += 32) {
      __syncthreads();
      stage_tile(sA, A1, lda1, m0, k0, K1, tid);
      __syncthreads();
      v16bf a = load_fragA_lds(sA, lane);
      acc = wmma_bf16(a, load_fragB_w(Wt, ldw, n0, k0, lane), acc);
    }
  }
  if (A2) {
    for (int k0 = 0; k0 < K2; k0 += 32) {
      __syncthreads();
      stage_tile(sA, A2, lda2, m0, k0, K2, tid);
      __syncthreads();
      v16bf a = load_fragA_lds(sA, lane);
      acc = wmma_bf16(a, load_fragB_w(Wt, ldw, n0, K1 + k0, lane), acc);
    }
  }

  const int col = n0 + (lane & 15);
  const int half = lane >> 4;
  const float bb = bias[col];
#pragma unroll
  for (int j = 0; j < 8; ++j) {
    long b = m0 + j + half * 8;
    Outb[b * ldo + col] = (__bf16)(acc[j] + bb);
  }
}

// ---------------------------------------------------------------------------
// Qnet head: mu/lv/z sample + 3 planar flows block-locally in LDS
// grid = (Bsz/16), block 256
// ---------------------------------------------------------------------------
__global__ __launch_bounds__(256) void qnet_z_kernel(
    const __bf16* __restrict__ Dq,                  // [Bsz x Dd] bf16
    const __bf16* __restrict__ Wmu, const __bf16* __restrict__ Wsg,
    const float* __restrict__ bmu, const float* __restrict__ bsg,
    const __bf16* __restrict__ Wp, const float* __restrict__ bp,
    const float* __restrict__ up,
    const float* __restrict__ eps, int t,
    float* __restrict__ out_mu, float* __restrict__ out_lv,
    __bf16* __restrict__ z_out)                     // [Bsz x Zd] bf16 step slice
{
  __shared__ __align__(16) __bf16 sA[16 * 32];
  __shared__ float sZ[16 * 128];
  const int tid = threadIdx.x, lane = tid & 31, wv = tid >> 5;
  const int m0 = blockIdx.x * 16;
  const int cw = wv * 16;
  const int half = lane >> 4;
  const int col = cw + (lane & 15);

  v8f amu{}, asg{};
  for (int k0 = 0; k0 < Dd; k0 += 32) {
    __syncthreads();
    stage_tile(sA, Dq, Dd, m0, k0, Dd, tid);
    __syncthreads();
    v16bf a = load_fragA_lds(sA, lane);
    amu = wmma_bf16(a, load_fragB_w(Wmu, Dd, cw, k0, lane), amu);
    asg = wmma_bf16(a, load_fragB_w(Wsg, Dd, cw, k0, lane), asg);
  }

  const float bm = bmu[col], bs = bsg[col];
#pragma unroll
  for (int j = 0; j < 8; ++j) {
    int row = j + half * 8;
    long b = m0 + row;
    float mu = amu[j] + bm;
    float lv = softplusf_(asg[j] + bs);
    float e  = __builtin_nontemporal_load(&eps[(b * Wt_ + t) * Zd + col]);
    float z  = mu + __expf(0.5f * lv) * e;
    __builtin_nontemporal_store(mu, &out_mu[(b * Wt_ + t) * Zd + col]);
    __builtin_nontemporal_store(lv, &out_lv[(b * Wt_ + t) * Zd + col]);
    sZ[row * 128 + col] = z;
  }
  __syncthreads();

  // planar flows: z = z + up[k] * tanh(z @ Wp[k].T + bp[k])
  for (int k = 0; k < Kfl; ++k) {
    const __bf16* Wk = Wp + k * Zd * Zd;
    v8f acc{};
    const int m = lane & 15;
    for (int kc = 0; kc < Zd; kc += 32) {
      v16bf a{};
      const float* zr = &sZ[m * 128 + kc + half * 8];
#pragma unroll
      for (int i = 0; i < 8; ++i) {
        a[i]     = (__bf16)zr[i];
        a[8 + i] = (__bf16)zr[16 + i];
      }
      acc = wmma_bf16(a, load_fragB_w(Wk, Zd, cw, kc, lane), acc);
    }
    float uk = up[k];
    float bk = bp[k * Zd + col];
    __syncthreads();               // all reads of sZ finished
#pragma unroll
    for (int j = 0; j < 8; ++j) {
      int row = j + half * 8;
      float z = sZ[row * 128 + col];
      z += uk * tanhf(acc[j] + bk);
      sZ[row * 128 + col] = z;
    }
    __syncthreads();
  }

#pragma unroll
  for (int j = 0; j < 8; ++j) {
    int row = j + half * 8;
    z_out[(long)(m0 + row) * Zd + col] = (__bf16)sZ[row * 128 + col];
  }
}

// ---------------------------------------------------------------------------
// Pnet head: out = (dd@Wmu.T+b) + softplus(dd@Wsig.T+b)*eps_p
// grid = (Bsz/16), block 256
// ---------------------------------------------------------------------------
__global__ __launch_bounds__(256) void pnet_out_kernel(
    const __bf16* __restrict__ dd,
    const __bf16* __restrict__ Wmu, const __bf16* __restrict__ Wsg,
    const float* __restrict__ bmu, const float* __restrict__ bsg,
    const float* __restrict__ eps, int t,
    float* __restrict__ out)
{
  __shared__ __align__(16) __bf16 sA[16 * 32];
  const int tid = threadIdx.x, lane = tid & 31, wv = tid >> 5;
  const int m0 = blockIdx.x * 16;
  const int cw = wv * 16;
  const int half = lane >> 4;
  const int col = cw + (lane & 15);

  v8f amu{}, asg{};
  for (int k0 = 0; k0 < Dd; k0 += 32) {
    __syncthreads();
    stage_tile(sA, dd, Dd, m0, k0, Dd, tid);
    __syncthreads();
    v16bf a = load_fragA_lds(sA, lane);
    amu = wmma_bf16(a, load_fragB_w(Wmu, Dd, cw, k0, lane), amu);
    asg = wmma_bf16(a, load_fragB_w(Wsg, Dd, cw, k0, lane), asg);
  }

  const float bm = bmu[col], bs = bsg[col];
#pragma unroll
  for (int j = 0; j < 8; ++j) {
    long b = m0 + j + half * 8;
    float mu = amu[j] + bm;
    float sg = softplusf_(asg[j] + bs);
    float e  = __builtin_nontemporal_load(&eps[(b * Wt_ + t) * OUTd + col]);
    __builtin_nontemporal_store(mu + sg * e, &out[(b * Wt_ + t) * OUTd + col]);
  }
}

// ---------------------------------------------------------------------------
// Host orchestration
// ---------------------------------------------------------------------------
extern "C" void kernel_launch(void* const* d_in, const int* in_sizes, int n_in,
                              void* d_out, int out_size, void* d_ws, size_t ws_size,
                              hipStream_t stream) {
  (void)in_sizes; (void)n_in; (void)out_size; (void)ws_size;

  const float* x      = (const float*)d_in[0];
  const float* eps_q  = (const float*)d_in[1];
  const float* eps_p  = (const float*)d_in[2];
  const float* Wih_q  = (const float*)d_in[3];
  const float* Whh_q  = (const float*)d_in[4];
  const float* bih_q  = (const float*)d_in[5];
  const float* bhh_q  = (const float*)d_in[6];
  const float* Wd_q   = (const float*)d_in[7];
  const float* bd_q   = (const float*)d_in[8];
  const float* Wmu_q  = (const float*)d_in[9];
  const float* bmu_q  = (const float*)d_in[10];
  const float* Wsig_q = (const float*)d_in[11];
  const float* bsig_q = (const float*)d_in[12];
  const float* Wp     = (const float*)d_in[13];
  const float* bp     = (const float*)d_in[14];
  const float* up     = (const float*)d_in[15];
  const float* Wlg    = (const float*)d_in[16];
  const float* blg    = (const float*)d_in[17];
  const float* Wih_p  = (const float*)d_in[18];
  const float* Whh_p  = (const float*)d_in[19];
  const float* bih_p  = (const float*)d_in[20];
  const float* bhh_p  = (const float*)d_in[21];
  const float* Wd_p   = (const float*)d_in[22];
  const float* bd_p   = (const float*)d_in[23];
  const float* Wmu_p  = (const float*)d_in[24];
  const float* bmu_p  = (const float*)d_in[25];
  const float* Wsig_p = (const float*)d_in[26];
  const float* bsig_p = (const float*)d_in[27];

  float* out_seq = (float*)d_out;                               // [B,W,OUT]
  float* out_mu  = out_seq + (long)Bsz * Wt_ * OUTd;            // [B,W,Z]
  float* out_lv  = out_mu  + (long)Bsz * Wt_ * Zd;              // [B,W,Z]

  // --- workspace bump allocator (256B aligned) ---
  char* wsb = (char*)d_ws;
  size_t off = 0;
  auto alloc = [&](size_t bytes) -> char* {
    char* p = wsb + off;
    off = (off + bytes + 255) & ~(size_t)255;
    return p;
  };

  __bf16* bWih_q  = (__bf16*)alloc((size_t)3 * Hd * INd * 2);
  __bf16* bWhh_q  = (__bf16*)alloc((size_t)3 * Hd * Hd * 2);
  __bf16* bWd_q   = (__bf16*)alloc((size_t)Dd * (Zd + Hd) * 2);
  __bf16* bWmu_q  = (__bf16*)alloc((size_t)Zd * Dd * 2);
  __bf16* bWsig_q = (__bf16*)alloc((size_t)Zd * Dd * 2);
  __bf16* bWp     = (__bf16*)alloc((size_t)Kfl * Zd * Zd * 2);
  __bf16* bWlg    = (__bf16*)alloc((size_t)Zd * Zd * 2);
  __bf16* bWih_p  = (__bf16*)alloc((size_t)3 * Hd * Zd * 2);
  __bf16* bWhh_p  = (__bf16*)alloc((size_t)3 * Hd * Hd * 2);
  __bf16* bWd_p   = (__bf16*)alloc((size_t)Dd * Hd * 2);
  __bf16* bWmu_p  = (__bf16*)alloc((size_t)OUTd * Dd * 2);
  __bf16* bWsig_p = (__bf16*)alloc((size_t)OUTd * Dd * 2);

  __bf16* xb      = (__bf16*)alloc((size_t)Bsz * Wt_ * INd * 2);   // bf16 copy of x
  __bf16* e0b     = (__bf16*)alloc((size_t)Bsz * Hd * 2);
  __bf16* e1b     = (__bf16*)alloc((size_t)Bsz * Hd * 2);
  __bf16* dp0b    = (__bf16*)alloc((size_t)Bsz * Hd * 2);
  __bf16* dp1b    = (__bf16*)alloc((size_t)Bsz * Hd * 2);
  __bf16* dbufb   = (__bf16*)alloc((size_t)Bsz * Dd * 2);
  __bf16* z_seqb  = (__bf16*)alloc((size_t)Wt_ * Bsz * Zd * 2);    // [t][b][z]
  __bf16* zlgb    = (__bf16*)alloc((size_t)Wt_ * Bsz * Zd * 2);    // [t][b][z]

  float* e0  = (float*)alloc((size_t)Bsz * Hd * 4);
  float* e1  = (float*)alloc((size_t)Bsz * Hd * 4);
  float* dp0 = (float*)alloc((size_t)Bsz * Hd * 4);
  float* dp1 = (float*)alloc((size_t)Bsz * Hd * 4);

  auto cvt = [&](const float* s, __bf16* d, long n) {
    cvt_bf16_kernel<<<dim3((unsigned)((n + 255) / 256)), 256, 0, stream>>>(s, d, (int)n);
  };
  cvt(Wih_q,  bWih_q,  (long)3 * Hd * INd);
  cvt(Whh_q,  bWhh_q,  (long)3 * Hd * Hd);
  cvt(Wd_q,   bWd_q,   (long)Dd * (Zd + Hd));
  cvt(Wmu_q,  bWmu_q,  (long)Zd * Dd);
  cvt(Wsig_q, bWsig_q, (long)Zd * Dd);
  cvt(Wp,     bWp,     (long)Kfl * Zd * Zd);
  cvt(Wlg,    bWlg,    (long)Zd * Zd);
  cvt(Wih_p,  bWih_p,  (long)3 * Hd * Zd);
  cvt(Whh_p,  bWhh_p,  (long)3 * Hd * Hd);
  cvt(Wd_p,   bWd_p,   (long)Dd * Hd);
  cvt(Wmu_p,  bWmu_p,  (long)OUTd * Dd);
  cvt(Wsig_p, bWsig_p, (long)OUTd * Dd);
  cvt(x,      xb,      (long)Bsz * Wt_ * INd);

  // zero initial states (f32 and bf16: 0x0000 == 0.0bf16)
  (void)hipMemsetAsync(e0,   0, (size_t)Bsz * Hd * 4, stream);
  (void)hipMemsetAsync(dp0,  0, (size_t)Bsz * Hd * 4, stream);
  (void)hipMemsetAsync(e0b,  0, (size_t)Bsz * Hd * 2, stream);
  (void)hipMemsetAsync(dp0b, 0, (size_t)Bsz * Hd * 2, stream);

  float*  ebuf[2]   = {e0, e1};
  __bf16* ebufb[2]  = {e0b, e1b};
  float*  dpbuf[2]  = {dp0, dp1};
  __bf16* dpbufb[2] = {dp0b, dp1b};

  // ---- Qnet: sequential encoder ----
  for (int t = 0; t < Wt_; ++t) {
    const __bf16* Xt = xb + (long)t * INd;             // row b at +b*(W*IN)
    gru_step_kernel<<<dim3(Bsz / 16, Hd / 128), 256, 0, stream>>>(
        Xt, (long)Wt_ * INd, ebuf[t & 1], ebufb[t & 1],
        bWih_q, bWhh_q, bih_q, bhh_q,
        ebuf[(t + 1) & 1], ebufb[(t + 1) & 1]);

    const __bf16* zprev = t ? (z_seqb + (long)(t - 1) * Bsz * Zd) : nullptr;
    gemm_bias_kernel<<<dim3(Bsz / 16, Dd / 128), 256, 0, stream>>>(
        zprev, (long)Zd, Zd,
        ebufb[(t + 1) & 1], (long)Hd, Hd,
        bWd_q, Zd + Hd, bd_q, dbufb, Dd);

    qnet_z_kernel<<<dim3(Bsz / 16), 256, 0, stream>>>(
        dbufb, bWmu_q, bWsig_q, bmu_q, bsig_q, bWp, bp, up,
        eps_q, t, out_mu, out_lv, z_seqb + (long)t * Bsz * Zd);
  }

  // ---- zlg = z_seq @ Wlg.T + blg over all W*B rows ----
  gemm_bias_kernel<<<dim3((Wt_ * Bsz) / 16, 1), 256, 0, stream>>>(
      z_seqb, (long)Zd, Zd, nullptr, 0, 0, bWlg, Zd, blg, zlgb, Zd);

  // ---- Pnet: sequential decoder ----
  for (int t = 0; t < Wt_; ++t) {
    gru_step_kernel<<<dim3(Bsz / 16, Hd / 128), 256, 0, stream>>>(
        zlgb + (long)t * Bsz * Zd, (long)Zd, dpbuf[t & 1], dpbufb[t & 1],
        bWih_p, bWhh_p, bih_p, bhh_p,
        dpbuf[(t + 1) & 1], dpbufb[(t + 1) & 1]);

    gemm_bias_kernel<<<dim3(Bsz / 16, Dd / 128), 256, 0, stream>>>(
        dpbufb[(t + 1) & 1], (long)Hd, Hd, nullptr, 0, 0,
        bWd_p, Hd, bd_p, dbufb, Dd);

    pnet_out_kernel<<<dim3(Bsz / 16), 256, 0, stream>>>(
        dbufb, bWmu_p, bWsig_p, bmu_p, bsig_p, eps_p, t, out_seq);
  }
}
